// Model_37091337568619
// MI455X (gfx1250) — matrix operand order repaired and checked by reading, not compiled, not run on hardware
//
#include <hip/hip_runtime.h>

// MI455X / gfx1250, wave32. Heavy GEMMs via v_wmma_f32_16x16x32_f16 with
// double-buffered LDS tiles staged by GLOBAL_LOAD_ASYNC_TO_LDS_B128 (ASYNCcnt).
//
// Algebraic collapse used (exact up to fp rounding):
//   single-KV attention + softmax => attn(s,i) = pe[:,i] @ W_attn + b_attn,
//   W_attn = (blockwise rot_w @ v_w) @ (sum of 15 row-blocks of qkv_w).
// This removes q_w, bias_layer and softmax entirely (~70% of reference FLOPs).

typedef _Float16 half_t;
typedef __attribute__((ext_vector_type(16))) _Float16 v16h_t;
typedef __attribute__((ext_vector_type(8)))  _Float16 v8h_t;
typedef __attribute__((ext_vector_type(8)))  float    v8f_t;

#define B_SZ 2048
#define FF_SZ 1024

#define CDNA5_ASYNC_LDS __has_builtin(__builtin_amdgcn_global_load_async_to_lds_b128)

#if CDNA5_ASYNC_LDS
// The builtin expects: (AS1 int4* global_src, AS3 int4* lds_dst, imm offset, imm cpol)
// (clang diagnostic: parameter type 'int __vector(4) __device__ *')
typedef int v4i_t __attribute__((vector_size(16)));
typedef __attribute__((address_space(1))) v4i_t* gptr128_t;
typedef __attribute__((address_space(3))) v4i_t* lptr128_t;
#endif

__device__ __forceinline__ void copy16_to_lds(const half_t* g, half_t* l)
{
#if CDNA5_ASYNC_LDS
  // async DMA: global -> LDS, 16 bytes, tracked by ASYNCcnt
  __builtin_amdgcn_global_load_async_to_lds_b128((gptr128_t)g, (lptr128_t)l, 0, 0);
#else
  *(v8h_t*)l = *(const v8h_t*)g;
#endif
}

__device__ __forceinline__ void async_wait_all()
{
#if CDNA5_ASYNC_LDS
# if __has_builtin(__builtin_amdgcn_s_wait_asynccnt)
  __builtin_amdgcn_s_wait_asynccnt(0);
# else
  asm volatile("s_wait_asynccnt 0" ::: "memory");
# endif
#endif
}

__device__ __forceinline__ float relu6f(float x){ return fminf(fmaxf(x,0.f),6.f); }

// ---------------------------------------------------------------------------
// Generic f16 WMMA GEMM:  C[M,N] = A[M,K](row-major, row stride lda) * B[K,N]
// where B is provided pre-transposed as Bt[N,K] row-major.
// Optional epilogue: + bias[N] + addend[M,N].
// Block: 128 threads (4 waves), 64x64 tile, K-step 32, double-buffered LDS
// with async global->LDS staging overlapping the WMMA of the current tile.
// Fragment loads follow the CDNA5 ISA VGPR layouts:
//   A 16x32 f16: lane L (M=L%16) holds K in {b..b+7, b+16..b+23}, b=8*(L>=16)
//   B 32x16 f16: lane L (N=L%16) holds K in {b..b+7, b+16..b+23}, b=8*(L>=16)
//   D 16x16 f32: lane L (N=L%16) holds rows r+8*(L>=16), r=0..7
// ---------------------------------------------------------------------------
__global__ __launch_bounds__(128) void gemm_f16_wmma(
    const half_t* __restrict__ A, long lda,
    const half_t* __restrict__ Bt,
    float* __restrict__ C,
    const float* __restrict__ bias,
    const float* __restrict__ addend,
    int M, int N, int K)
{
  __shared__ __align__(16) half_t sA[2][64][32];
  __shared__ __align__(16) half_t sB[2][64][32];
  const int bm = blockIdx.y * 64, bn = blockIdx.x * 64;
  const int tid = threadIdx.x;
  const int wave = tid >> 5, lane = tid & 31;
  const int wm = (wave >> 1) * 32, wn = (wave & 1) * 32;
  const int r16 = lane & 15;
  const int hiA = (lane >> 4) << 3;   // 0 or 8
  const int hiB = (lane >> 4) << 3;   // 0 or 8

  v8f_t zero = {};
  v8f_t acc[2][2];
  acc[0][0]=zero; acc[0][1]=zero; acc[1][0]=zero; acc[1][1]=zero;

  auto stage = [&](int kt, int bf) {
    const int k0 = kt << 5;
    #pragma unroll
    for (int c = 0; c < 2; ++c) {
      int e  = tid + c*128;             // 0..255 chunk id
      int r  = e >> 2;
      int cc = (e & 3) << 3;
      copy16_to_lds(&A[(long)(bm + r)*lda + k0 + cc],          &sA[bf][r][cc]);
      copy16_to_lds(&Bt[(long)(bn + r)*(long)K + k0 + cc],     &sB[bf][r][cc]);
    }
  };

  const int nk = K >> 5;
  stage(0, 0);
  async_wait_all();
  __syncthreads();

  for (int kt = 0; kt < nk; ++kt) {
    const int cur = kt & 1;
    if (kt + 1 < nk) {
      stage(kt + 1, cur ^ 1);           // async fill of the other buffer
      if (kt + 2 < nk) {                // L2 prefetch two tiles ahead
        int r = tid >> 1;
        __builtin_prefetch(&A[(long)(bm + r)*lda + ((kt+2) << 5)], 0, 3);
        __builtin_prefetch(&Bt[(long)(bn + r)*(long)K + ((kt+2) << 5)], 0, 3);
      }
    }

    v16h_t a[2], b[2];
    #pragma unroll
    for (int mt = 0; mt < 2; ++mt) {
      const half_t* p = &sA[cur][wm + mt*16 + r16][hiA];
      v8h_t lo = *(const v8h_t*)p;          // K = hiA .. hiA+7
      v8h_t hi = *(const v8h_t*)(p + 16);   // K = hiA+16 .. hiA+23
      a[mt] = __builtin_shufflevector(lo, hi, 0,1,2,3,4,5,6,7,8,9,10,11,12,13,14,15);
    }
    #pragma unroll
    for (int nt = 0; nt < 2; ++nt) {
      const half_t* p = &sB[cur][wn + nt*16 + r16][hiB];
      v8h_t lo = *(const v8h_t*)p;          // K = hiB .. hiB+7
      v8h_t hi = *(const v8h_t*)(p + 16);   // K = hiB+16 .. hiB+23
      b[nt] = __builtin_shufflevector(lo, hi, 0,1,2,3,4,5,6,7,8,9,10,11,12,13,14,15);
    }
    #pragma unroll
    for (int mt = 0; mt < 2; ++mt)
      #pragma unroll
      for (int nt = 0; nt < 2; ++nt)
        acc[mt][nt] = __builtin_amdgcn_wmma_f32_16x16x32_f16(
            false, a[mt], false, b[nt], (short)0, acc[mt][nt], false, false);

    async_wait_all();                   // next buffer fully in LDS
    __syncthreads();                    // all waves done reading 'cur'
  }

  const int colL = lane & 15, rbase = (lane >> 4) << 3;
  #pragma unroll
  for (int mt = 0; mt < 2; ++mt)
    #pragma unroll
    for (int nt = 0; nt < 2; ++nt) {
      int col = bn + wn + nt*16 + colL;
      float bv = bias ? bias[col] : 0.f;
      #pragma unroll
      for (int r = 0; r < 8; ++r) {
        int row = bm + wm + mt*16 + rbase + r;
        float v = acc[mt][nt][r] + bv;
        if (addend) v += addend[(long)row*N + col];
        C[(long)row*N + col] = v;
      }
    }
}

// ---------------------------------------------------------------------------
// Fused residual-add + LayerNorm over 1024 cols; optional f16 shadow copy.
// ---------------------------------------------------------------------------
__global__ __launch_bounds__(256) void add_ln_kernel(
    const float* __restrict__ X, const float* __restrict__ R,
    const float* __restrict__ g, const float* __restrict__ bta,
    float* __restrict__ Y, half_t* __restrict__ Yh)
{
  __shared__ float r1[256], r2[256];
  const int b = blockIdx.x, t = threadIdx.x;
  const float* x = X + (long)b*FF_SZ;
  const float* r = R + (long)b*FF_SZ;
  float v[4]; float a0 = 0.f, a1 = 0.f;
  #pragma unroll
  for (int i = 0; i < 4; ++i) {
    int idx = t + 256*i;
    float u = x[idx] + r[idx];
    v[i] = u; a0 += u; a1 += u*u;
  }
  r1[t] = a0; r2[t] = a1; __syncthreads();
  for (int off = 128; off > 0; off >>= 1) {
    if (t < off) { r1[t] += r1[t+off]; r2[t] += r2[t+off]; }
    __syncthreads();
  }
  float mean = r1[0] * (1.f/FF_SZ);
  float var  = r2[0] * (1.f/FF_SZ) - mean*mean;
  float inv  = rsqrtf(var + 1e-5f);
  #pragma unroll
  for (int i = 0; i < 4; ++i) {
    int idx = t + 256*i;
    float o = (v[i] - mean) * inv * g[idx] + bta[idx];
    Y[(long)b*FF_SZ + idx] = o;
    if (Yh) Yh[(long)b*FF_SZ + idx] = (half_t)o;
  }
}

// ---------------------------------------------------------------------------
// Fused CNN tower: conv3x3(1->3) -> 1x1(3->64) -> bn+relu6+pool2 ->
// depthwise3x3 -> 1x1(64->512) -> bn+relu6 -> (pad,pool2) => feat[B,7680] f16.
// One block per batch element, everything staged in LDS.
// (Note: the W-pad column and bottom row are dropped by the VALID 2x2 pool.)
// ---------------------------------------------------------------------------
__global__ __launch_bounds__(256) void cnn_kernel(
  const float* __restrict__ board,
  const float* __restrict__ c1w, const float* __restrict__ c1b,
  const float* __restrict__ p1w, const float* __restrict__ p1b,
  const float* __restrict__ bn1g, const float* __restrict__ bn1b,
  const float* __restrict__ bn1m, const float* __restrict__ bn1v,
  const float* __restrict__ c2w, const float* __restrict__ c2b,
  const float* __restrict__ p2w, const float* __restrict__ p2b,
  const float* __restrict__ bn2g, const float* __restrict__ bn2b,
  const float* __restrict__ bn2m, const float* __restrict__ bn2v,
  half_t* __restrict__ feat)
{
  __shared__ float sb[24*14];
  __shared__ float o1[3*22*12];
  __shared__ float t1[64*11*6];
  __shared__ float t2[64*11*6];
  const int b = blockIdx.x, t = threadIdx.x;
  const float* bd = board + (long)b*336;
  for (int i = t; i < 336; i += 256) sb[i] = bd[i];
  __syncthreads();
  for (int i = t; i < 792; i += 256) {          // conv1 VALID: (3,22,12)
    int cc = i/264, p = i%264, ii = p/12, jj = p%12;
    float acc = c1b[cc];
    #pragma unroll
    for (int u = 0; u < 3; ++u)
      #pragma unroll
      for (int v = 0; v < 3; ++v)
        acc += sb[(ii+u)*14 + jj+v] * c1w[cc*9 + u*3 + v];
    o1[i] = acc;
  }
  __syncthreads();
  for (int i = t; i < 4224; i += 256) {         // p1 + bn1 + relu6 + pool -> (64,11,6)
    int c = i/66, p = i%66, ih = p/6, iw = p%6;
    float sc = bn1g[c]*rsqrtf(bn1v[c]+1e-5f);
    float sh = bn1b[c] - bn1m[c]*sc;
    float m = -1e30f;
    #pragma unroll
    for (int dy = 0; dy < 2; ++dy)
      #pragma unroll
      for (int dx = 0; dx < 2; ++dx) {
        int ii = 2*ih+dy, jj = 2*iw+dx;
        float acc = p1b[c];
        #pragma unroll
        for (int cc = 0; cc < 3; ++cc) acc += o1[cc*264 + ii*12 + jj] * p1w[c*3 + cc];
        m = fmaxf(m, relu6f(acc*sc + sh));
      }
    t1[i] = m;
  }
  __syncthreads();
  for (int i = t; i < 4224; i += 256) {         // depthwise 3x3, zero pad
    int c = i/66, p = i%66, ih = p/6, iw = p%6;
    float acc = c2b[c];
    #pragma unroll
    for (int u = 0; u < 3; ++u)
      #pragma unroll
      for (int v = 0; v < 3; ++v) {
        int yy = ih+u-1, xx = iw+v-1;
        if (yy >= 0 && yy < 11 && xx >= 0 && xx < 6)
          acc += t1[c*66 + yy*6 + xx] * c2w[c*9 + u*3 + v];
      }
    t2[i] = acc;
  }
  __syncthreads();
  for (int i = t; i < 7680; i += 256) {         // p2 + bn2 + relu6 + pool -> (512,5,3)
    int oc = i/15, pos = i%15, oh = pos/3, ow = pos%3;
    float sc = bn2g[oc]*rsqrtf(bn2v[oc]+1e-5f);
    float sh = bn2b[oc] - bn2m[oc]*sc;
    const float* wrow = p2w + (long)oc*64;
    float m = -1e30f;
    for (int dy = 0; dy < 2; ++dy)
      for (int dx = 0; dx < 2; ++dx) {
        int h = 2*oh+dy, w2 = 2*ow+dx;
        float acc = p2b[oc];
        #pragma unroll 8
        for (int c = 0; c < 64; ++c) acc += t2[c*66 + h*6 + w2] * wrow[c];
        m = fmaxf(m, relu6f(acc*sc + sh));
      }
    feat[(long)b*7680 + i] = (half_t)m;
  }
}

// ---------------------------------------------------------------------------
// Small / precompute kernels
// ---------------------------------------------------------------------------
__global__ __launch_bounds__(256) void pe_kernel(        // pe[B,7,512] -> f16
  const float* __restrict__ xyrot, const int* __restrict__ queue,
  const float* __restrict__ locw, const float* __restrict__ locb,
  const float* __restrict__ emb, half_t* __restrict__ pe16)
{
  long idx = (long)blockIdx.x*256 + threadIdx.x;   // B*7*512
  int col = (int)(idx % 512);
  long row = idx / 512;
  int p = (int)(row % 7);
  long b = row / 7;
  int q = queue[b*7 + p];
  float v = emb[(long)q*512 + col];
  if (p == 0)
    v += locb[col] + xyrot[b*3+0]*locw[col] + xyrot[b*3+1]*locw[512+col]
       + xyrot[b*3+2]*locw[1024+col];
  pe16[idx] = (half_t)v;
}

__global__ __launch_bounds__(256) void gp_kernel(        // garbage path -> gp[B,1024]
  const float* __restrict__ garbage, const float* __restrict__ gcw,
  const float* __restrict__ gcb, const float* __restrict__ g2w,
  const float* __restrict__ g2b, float* __restrict__ gp)
{
  long idx = (long)blockIdx.x*256 + threadIdx.x;   // B*1024
  int f = (int)(idx % 1024); long b = idx / 1024;
  float gv = garbage[b];
  float acc = g2b[f];
  #pragma unroll 8
  for (int c = 0; c < 64; ++c)
    acc += (gcw[c*3+1]*gv + gcb[c]) * g2w[(long)c*1024 + f];
  gp[idx] = acc;
}

__global__ __launch_bounds__(256) void sumqkv_kernel(    // qs[512,1024]
  const float* __restrict__ qkvw, float* __restrict__ qs)
{
  long idx = (long)blockIdx.x*256 + threadIdx.x;   // 512*1024
  int f = (int)(idx % 1024); int r = (int)(idx / 1024);
  float a = 0.f;
  #pragma unroll
  for (int j = 0; j < 15; ++j) a += qkvw[((long)j*512 + r)*1024 + f];
  qs[idx] = a;
}

__global__ __launch_bounds__(256) void wvf_kernel(       // Wvf[512,512]
  const float* __restrict__ rotw, const float* __restrict__ vw, float* __restrict__ wvf)
{
  long idx = (long)blockIdx.x*256 + threadIdx.x;   // 512*512
  int o = (int)(idx % 512); int r = (int)(idx / 512);
  int j = o/128, c = o%128;
  const float* rw = rotw + (long)r*2048 + j*512;
  float a = 0.f;
  for (int k = 0; k < 512; ++k) a += rw[k] * vw[(long)k*128 + c];
  wvf[idx] = a;
}

__global__ __launch_bounds__(256) void bvf_kernel(       // bvf[512]
  const float* __restrict__ rotb, const float* __restrict__ vw,
  const float* __restrict__ vb, float* __restrict__ bvf)
{
  int o = blockIdx.x*256 + threadIdx.x; if (o >= 512) return;
  int j = o/128, c = o%128;
  float a = vb[c];
  for (int k = 0; k < 512; ++k) a += rotb[j*512 + k] * vw[(long)k*128 + c];
  bvf[o] = a;
}

__global__ __launch_bounds__(256) void wattn_kernel(     // WattnT[1024,512] f16
  const float* __restrict__ wvf, const float* __restrict__ qs, half_t* __restrict__ wattnT)
{
  long idx = (long)blockIdx.x*256 + threadIdx.x;   // 512*1024
  int f = (int)(idx % 1024); int r = (int)(idx / 1024);
  float a = 0.f;
  for (int m = 0; m < 512; ++m) a += wvf[(long)r*512 + m] * qs[(long)m*1024 + f];
  wattnT[(long)f*512 + r] = (half_t)a;
}

__global__ __launch_bounds__(256) void battn_kernel(     // battn[1024]
  const float* __restrict__ bvf, const float* __restrict__ qs,
  const float* __restrict__ qkvb, float* __restrict__ battn)
{
  int f = blockIdx.x*256 + threadIdx.x; if (f >= 1024) return;
  float a = qkvb[f];
  for (int r = 0; r < 512; ++r) a += bvf[r] * qs[(long)r*1024 + f];
  battn[f] = a;
}

__global__ __launch_bounds__(256) void convT_kernel(     // W[K,N] f32 -> Wt[N,K] f16
  const float* __restrict__ src, half_t* __restrict__ dst, int K, int N)
{
  long idx = (long)blockIdx.x*256 + threadIdx.x;
  if (idx >= (long)K*N) return;
  int n = (int)(idx % N); long k = idx / N;
  dst[(long)n*K + k] = (half_t)src[idx];
}

// ---------------------------------------------------------------------------
extern "C" void kernel_launch(void* const* d_in, const int* in_sizes, int n_in,
                              void* d_out, int out_size, void* d_ws, size_t ws_size,
                              hipStream_t stream)
{
  (void)in_sizes; (void)n_in; (void)out_size; (void)ws_size;
  const float* xyrot    = (const float*)d_in[0];
  const int*   queue    = (const int*)  d_in[1];
  const float* board    = (const float*)d_in[3];
  const float* garbage  = (const float*)d_in[4];
  const float* loc_w    = (const float*)d_in[5];
  const float* loc_b    = (const float*)d_in[6];
  const float* piece_emb= (const float*)d_in[7];
  const float* rot_w    = (const float*)d_in[8];
  const float* rot_b    = (const float*)d_in[9];
  const float* gconv_w  = (const float*)d_in[10];
  const float* gconv_b  = (const float*)d_in[11];
  const float* g2_w     = (const float*)d_in[12];
  const float* g2_b     = (const float*)d_in[13];
  const float* c1_w     = (const float*)d_in[14];
  const float* c1_b     = (const float*)d_in[15];
  const float* p1_w     = (const float*)d_in[16];
  const float* p1_b     = (const float*)d_in[17];
  const float* bn1_g    = (const float*)d_in[18];
  const float* bn1_b    = (const float*)d_in[19];
  const float* bn1_m    = (const float*)d_in[20];
  const float* bn1_v    = (const float*)d_in[21];
  const float* c2_w     = (const float*)d_in[22];
  const float* c2_b     = (const float*)d_in[23];
  const float* p2_w     = (const float*)d_in[24];
  const float* p2_b     = (const float*)d_in[25];
  const float* bn2_g    = (const float*)d_in[26];
  const float* bn2_b    = (const float*)d_in[27];
  const float* bn2_m    = (const float*)d_in[28];
  const float* bn2_v    = (const float*)d_in[29];
  const float* cnn_w    = (const float*)d_in[30];
  const float* cnn_b    = (const float*)d_in[31];
  const float* v_w      = (const float*)d_in[36];
  const float* v_b      = (const float*)d_in[37];
  const float* qkv_w    = (const float*)d_in[38];
  const float* qkv_b    = (const float*)d_in[39];
  const float* queue_w  = (const float*)d_in[51];
  const float* queue_b  = (const float*)d_in[52];
  const float* remain_w = (const float*)d_in[53];
  const float* remain_b = (const float*)d_in[54];
  const float* attn_g   = (const float*)d_in[55];
  const float* attn_beta= (const float*)d_in[56];
  const float* ff_g     = (const float*)d_in[57];
  const float* ff_beta  = (const float*)d_in[58];

  char* w = (char*)d_ws;
  auto carve = [&](size_t bytes) -> void* {
    void* p = (void*)w; w += (bytes + 255) & ~(size_t)255; return p;
  };
  float*  qs     = (float*) carve(512l*1024*4);
  float*  wvf    = (float*) carve(512l*512*4);
  float*  bvf    = (float*) carve(512*4);
  half_t* wattnT = (half_t*)carve(1024l*512*2);
  float*  battn  = (float*) carve(1024*4);
  half_t* cnnT   = (half_t*)carve(1024l*7680*2);
  half_t* qwT    = (half_t*)carve(1024l*1024*2);
  half_t* rwT    = (half_t*)carve(1024l*1024*2);
  half_t* pe16   = (half_t*)carve((long)B_SZ*7*512*2);
  half_t* feat16 = (half_t*)carve((long)B_SZ*7680*2);
  float*  gp     = (float*) carve((long)B_SZ*1024*4);
  float*  S      = (float*) carve((long)B_SZ*1024*4);
  half_t* Sh     = (half_t*)carve((long)B_SZ*1024*2);
  float*  T      = (float*) carve((long)B_SZ*1024*4);

  // ---- one-time weight folding (f32 for precision; f16 only at the end) ----
  sumqkv_kernel<<<(512*1024)/256, 256, 0, stream>>>(qkv_w, qs);
  wvf_kernel   <<<(512*512)/256,  256, 0, stream>>>(rot_w, v_w, wvf);
  bvf_kernel   <<<2, 256, 0, stream>>>(rot_b, v_w, v_b, bvf);
  wattn_kernel <<<(512*1024)/256, 256, 0, stream>>>(wvf, qs, wattnT);
  battn_kernel <<<4, 256, 0, stream>>>(bvf, qs, qkv_b, battn);
  convT_kernel <<<(int)((7680l*1024+255)/256), 256, 0, stream>>>(cnn_w, cnnT, 7680, 1024);
  convT_kernel <<<(1024*1024)/256, 256, 0, stream>>>(queue_w, qwT, 1024, 1024);
  convT_kernel <<<(1024*1024)/256, 256, 0, stream>>>(remain_w, rwT, 1024, 1024);

  // ---- activations ----
  pe_kernel <<<(B_SZ*7*512)/256, 256, 0, stream>>>(xyrot, queue, loc_w, loc_b, piece_emb, pe16);
  gp_kernel <<<(B_SZ*1024)/256, 256, 0, stream>>>(garbage, gconv_w, gconv_b, g2_w, g2_b, gp);
  cnn_kernel<<<B_SZ, 256, 0, stream>>>(board, c1_w,c1_b, p1_w,p1_b, bn1_g,bn1_b,bn1_m,bn1_v,
                                       c2_w,c2_b, p2_w,p2_b, bn2_g,bn2_b,bn2_m,bn2_v, feat16);

  // state = feat @ cnn_w + cnn_b + gp
  gemm_f16_wmma<<<dim3(1024/64, B_SZ/64), 128, 0, stream>>>(
      feat16, 7680, cnnT, S, cnn_b, gp, B_SZ, 1024, 7680);

  float* outF = (float*)d_out;
  for (int layer = 0; layer < 8; ++layer) {
    int i = (layer < 7) ? layer : 6;
    // attn residual: pe[:,i] @ W_attn + b_attn   (A rows strided by 7*512)
    gemm_f16_wmma<<<dim3(16, B_SZ/64), 128, 0, stream>>>(
        pe16 + (long)i*512, 7*512, wattnT, T, battn, nullptr, B_SZ, 1024, 512);
    add_ln_kernel<<<B_SZ, 256, 0, stream>>>(S, T, attn_g, attn_beta, S, Sh);
    // FF residual: state @ W + b
    const half_t* Wt = (layer < 7) ? qwT : rwT;
    const float*  bb = (layer < 7) ? queue_b : remain_b;
    gemm_f16_wmma<<<dim3(16, B_SZ/64), 128, 0, stream>>>(
        Sh, 1024, Wt, T, bb, nullptr, B_SZ, 1024, 1024);
    bool last = (layer == 7);
    add_ln_kernel<<<B_SZ, 256, 0, stream>>>(S, T, ff_g, ff_beta,
                                            last ? outF : S, last ? nullptr : Sh);
  }
}
